// BiLstmCrfNetwork_34840774705259
// MI455X (gfx1250) — matrix operand order repaired and checked by reading, not compiled, not run on hardware
//
#include <hip/hip_runtime.h>

// ---------------------------------------------------------------------------
// BiLSTM-CRF for MI455X (gfx1250), wave32 + WMMA f16 (16x16x32, f32 accum),
// TDM (tensor_load_to_lds) double-buffered prefetch in the recurrence.
// T=2048, H=300, D=320 (pad-free: 320 = 10*32), 4H=1200 = 75*16.
// ---------------------------------------------------------------------------

typedef __attribute__((ext_vector_type(16))) _Float16 v16h;
typedef __attribute__((ext_vector_type(8)))  _Float16 v8h;
typedef __attribute__((ext_vector_type(8)))  float    v8f;
typedef __attribute__((ext_vector_type(4)))  unsigned int v4u;
typedef __attribute__((ext_vector_type(8)))  int      v8i;
typedef __attribute__((ext_vector_type(4)))  int      v4i;

#define T_LEN   2048
#define HID     300
#define DIN     320     // 300 emb + 20 feats
#define G4H     1200    // 4*H
#define NTAGS   12
#define START_T 10
#define STOP_T  11
#define NEGINF  -10000.0f
#define KP_TAG  608     // 600 padded to 19*32

__device__ __forceinline__ v16h mk16(v8h lo, v8h hi) {
    v16h r;
#pragma unroll
    for (int i = 0; i < 8; i++) { r[i] = lo[i]; r[i + 8] = hi[i]; }
    return r;
}

__device__ __forceinline__ float fsig(float x) {
    return 1.0f / (1.0f + __expf(-x));
}

// TDM: DMA one f32 row of length G4H from global into LDS (D# per ISA ch.8).
// This toolchain exposes the 6-arg builtin: (g0, g1, g2, g3, g_extra, cpol).
__device__ __forceinline__ void tdm_load_row(const float* gptr, unsigned int lds_off) {
    unsigned long long ga = (unsigned long long)(size_t)gptr;
    v4u g0;
    g0[0] = 1u;                                          // count=1, user mode
    g0[1] = lds_off;                                     // lds_addr [63:32]
    g0[2] = (unsigned int)(ga & 0xFFFFFFFFu);            // global_addr lo
    g0[3] = (unsigned int)((ga >> 32) & 0x01FFFFFFu)     // global_addr hi (57b)
          | 0x80000000u;                                 // type=2 ("image")
    v8i g1;
    g1[0] = (int)0x00020000;                             // data_size=2 (4B), mask=0
    g1[1] = (int)((G4H & 0xFFFF) << 16);                 // tensor_dim0[15:0] @ [63:48]
    g1[2] = (int)(1 << 16);                              // tensor_dim1[15:0]=1 @ [95:80]
    g1[3] = (int)(G4H << 16);                            // tile_dim0 @ [127:112]
    g1[4] = 1;                                           // tile_dim1=1 @ [143:128]
    g1[5] = (int)G4H;                                    // tensor_dim0_stride lo32
    g1[6] = 0;
    g1[7] = 0;
    v4i gz4 = {0, 0, 0, 0};
    v8i gz8 = {0, 0, 0, 0, 0, 0, 0, 0};
    __builtin_amdgcn_tensor_load_to_lds(g0, g1, gz4, gz4, gz8, 0);
}

// ---------------------------------------------------------------------------
// Prep kernels: gather + f32 -> f16 packing into WMMA-friendly layouts
// ---------------------------------------------------------------------------
__global__ void pack_x(const int* __restrict__ idx,
                       const float* __restrict__ feats20,
                       const float* __restrict__ emb,
                       _Float16* __restrict__ X16) {
    int t = blockIdx.x, j = threadIdx.x;                    // 2048 x 320
    float v = (j < 300) ? emb[(size_t)idx[t] * 300 + j]
                        : feats20[t * 20 + (j - 300)];
    X16[(size_t)t * DIN + j] = (_Float16)v;
}

__global__ void pack_wi(const float* __restrict__ Wi_f,
                        const float* __restrict__ Wi_b,
                        _Float16* __restrict__ Wi16) {
    int n = blockIdx.x, k = threadIdx.x;                    // 2400 x 320
    const float* W = (n < G4H) ? Wi_f : Wi_b;
    int nn = (n < G4H) ? n : n - G4H;
    Wi16[(size_t)n * DIN + k] = (_Float16)W[(size_t)nn * DIN + k];
}

__global__ void pack_wh(const float* __restrict__ Wh_f,
                        const float* __restrict__ Wh_b,
                        _Float16* __restrict__ Wh16) {
    int n = blockIdx.x, k = threadIdx.x;                    // 2400 x 320 (pad K 300->320)
    const float* W = (n < G4H) ? Wh_f : Wh_b;
    int nn = (n < G4H) ? n : n - G4H;
    _Float16 v = (_Float16)0.f;
    if (k < HID) v = (_Float16)W[(size_t)nn * HID + k];
    Wh16[(size_t)n * DIN + k] = v;
}

__global__ void pack_bias(const float* __restrict__ bi_f, const float* __restrict__ bh_f,
                          const float* __restrict__ bi_b, const float* __restrict__ bh_b,
                          float* __restrict__ bias) {
    int i = blockIdx.x * 256 + threadIdx.x;                 // 2400
    if (i < G4H)            bias[i] = bi_f[i] + bh_f[i];
    else if (i < 2 * G4H)   bias[i] = bi_b[i - G4H] + bh_b[i - G4H];
}

__global__ void pack_wtag(const float* __restrict__ W_tag,
                          _Float16* __restrict__ wtag16) {
    int n = blockIdx.x, k = threadIdx.x;                    // 16 x 608
    _Float16 v = (_Float16)0.f;
    if (n < NTAGS && k < 600) v = (_Float16)W_tag[n * 600 + k];
    wtag16[(size_t)n * KP_TAG + k] = v;
}

__global__ void zero_pad_lo(_Float16* __restrict__ lo16) {
    int t = blockIdx.x;                                     // 2048 x 8 (cols 600..607)
    lo16[(size_t)t * KP_TAG + 600 + threadIdx.x] = (_Float16)0.f;
}

// ---------------------------------------------------------------------------
// GEMM 1: preact[dir][t][n] = X[t,:] . Wi[dir][n,:] + (bi+bh)[dir][n]
// Block = 4 waves (128 thr); each wave owns one 16x16 output tile.
// ---------------------------------------------------------------------------
__global__ __launch_bounds__(128) void gemm_preact(const _Float16* __restrict__ X,
                                                   const _Float16* __restrict__ Wi,
                                                   const float* __restrict__ bias,
                                                   float* __restrict__ preact) {
    int lane = threadIdx.x & 31, wave = threadIdx.x >> 5, hi = lane >> 4;
    int m0 = blockIdx.x * 16;
    int nt = blockIdx.y * 4 + wave;
    int dir = blockIdx.z;
    if (nt >= 75) return;
    int n0 = nt * 16;

    const _Float16* W  = Wi + (size_t)dir * G4H * DIN;
    const _Float16* xr = X + (size_t)(m0 + (lane & 15)) * DIN;
    const _Float16* wr = W + (size_t)(n0 + (lane & 15)) * DIN + hi * 16;

    v8f acc = {};
#pragma unroll
    for (int k = 0; k < 10; k++) {
        int k0 = k * 32;
        v16h a = mk16(*(const v8h*)(xr + k0 + hi * 16),
                      *(const v8h*)(xr + k0 + hi * 16 + 8));
        v16h b = mk16(*(const v8h*)(wr + k0),
                      *(const v8h*)(wr + k0 + 8));
        acc = __builtin_amdgcn_wmma_f32_16x16x32_f16(false, a, false, b,
                                                     (short)0, acc, false, false);
    }
    int n = n0 + (lane & 15);
    float bb = bias[dir * G4H + n];
    float* out = preact + ((size_t)dir * T_LEN + m0) * G4H;
#pragma unroll
    for (int r = 0; r < 8; r++) {
        int m = r + 8 * hi;                 // D layout: vgpr r -> M=r (lanes 0-15), M=r+8 (16-31)
        out[(size_t)m * G4H + n] = acc[r] + bb;
    }
}

// ---------------------------------------------------------------------------
// Sequential LSTM recurrence: 1 workgroup per direction, 16 waves.
// Wh (f16, K-padded to 320) pinned in VGPRs as 5x10 B-fragments per wave.
// A fragments (h replicated in all 16 rows) re-read from LDS per tile to
// keep VGPR pressure under the 16-wave budget.
// Preactivation rows are TDM-prefetched into a double-buffered LDS slot one
// step ahead (TENSORcnt) so L2 latency is off the dependent chain.
// ---------------------------------------------------------------------------
__global__ __launch_bounds__(512) void lstm_rec(const _Float16* __restrict__ Wh16,
                                                const float* __restrict__ preact,
                                                const float* __restrict__ h0,
                                                const float* __restrict__ c0,
                                                _Float16* __restrict__ lo16) {
    __shared__ __align__(16) _Float16 h16[DIN];   // current h, f16, K-padded
    __shared__ __align__(16) float pbuf[2][G4H];  // TDM-prefetched preact rows
    __shared__ float g[G4H];                      // raw matvec results
    int tid = threadIdx.x, lane = tid & 31, wave = tid >> 5, hi = lane >> 4;
    int dir = blockIdx.x;

    if (tid < DIN) h16[tid] = (tid < HID) ? (_Float16)h0[dir * HID + tid] : (_Float16)0.f;
    float c = (tid < HID) ? c0[dir * HID + tid] : 0.f;

    // Pin B tiles of Wh in registers: wave w covers n-tiles {w, w+16, ..., w+64} (<75)
    const _Float16* W = Wh16 + (size_t)dir * G4H * DIN;
    v16h B[5][10];
#pragma unroll
    for (int i = 0; i < 5; i++) {
        int nt = wave + 16 * i;
        if (nt >= 75) nt = 0;                               // dummy (result discarded)
        const _Float16* wr = W + (size_t)(nt * 16 + (lane & 15)) * DIN + hi * 16;
#pragma unroll
        for (int k = 0; k < 10; k++)
            B[i][k] = mk16(*(const v8h*)(wr + k * 32),
                           *(const v8h*)(wr + k * 32 + 8));
    }

    const float* pbase = preact + (size_t)dir * T_LEN * G4H;
    unsigned int pbuf_off0 = (unsigned int)(size_t)(&pbuf[0][0]);
    unsigned int pbuf_off1 = (unsigned int)(size_t)(&pbuf[1][0]);

    // Prologue: prefetch row for s=0 into pbuf[0]
    if (wave == 0) {
        int t0 = dir ? (T_LEN - 1) : 0;
        tdm_load_row(pbase + (size_t)t0 * G4H, pbuf_off0);
    }
    __syncthreads();

    for (int s = 0; s < T_LEN; s++) {
        // Issue prefetch for next step (clamped re-fetch at the tail; harmless)
        if (wave == 0) {
            int sn = (s + 1 < T_LEN) ? s + 1 : s;
            int tn = dir ? (T_LEN - 1 - sn) : sn;
            tdm_load_row(pbase + (size_t)tn * G4H,
                         ((s + 1) & 1) ? pbuf_off1 : pbuf_off0);
        }

        // Matvec: h (replicated rows) x Wh-tiles via WMMA
#pragma unroll
        for (int i = 0; i < 5; i++) {
            v8f acc = {};
#pragma unroll
            for (int k = 0; k < 10; k++) {
                v16h a = mk16(*(const v8h*)(&h16[k * 32 + hi * 16]),
                              *(const v8h*)(&h16[k * 32 + hi * 16 + 8]));
                acc = __builtin_amdgcn_wmma_f32_16x16x32_f16(false, a, false, B[i][k],
                                                             (short)0, acc, false, false);
            }
            int nt = wave + 16 * i;
            if (nt < 75 && lane < 16) {                     // row M=0 lives in acc[0]
                g[nt * 16 + lane] = acc[0];
            }
        }
        // Ensure the preact row for THIS step (issued last iteration) landed.
        if (wave == 0) __builtin_amdgcn_s_wait_tensorcnt(1);
        __syncthreads();

        if (tid < HID) {
            const float* pr = pbuf[s & 1];
            float ig = fsig(g[tid]           + pr[tid]);
            float fg = fsig(g[HID + tid]     + pr[HID + tid]);
            float gg = tanhf(g[2 * HID + tid] + pr[2 * HID + tid]);
            float og = fsig(g[3 * HID + tid] + pr[3 * HID + tid]);
            c = fg * c + ig * gg;
            float h = og * tanhf(c);
            int t = dir ? (T_LEN - 1 - s) : s;
            lo16[(size_t)t * KP_TAG + dir * HID + tid] = (_Float16)h;   // [fw | bw]
            h16[tid] = (_Float16)h;
        }
        __syncthreads();
    }
}

// ---------------------------------------------------------------------------
// GEMM 2: feats[t][n] = lstm_out[t,:600] . W_tag[n,:600] + b_tag[n], N padded 16
// ---------------------------------------------------------------------------
__global__ __launch_bounds__(128) void gemm_tag(const _Float16* __restrict__ lo16,
                                                const _Float16* __restrict__ wtag16,
                                                const float* __restrict__ b_tag,
                                                float* __restrict__ feats) {
    int lane = threadIdx.x & 31, wave = threadIdx.x >> 5, hi = lane >> 4;
    int m0 = (blockIdx.x * 4 + wave) * 16;
    const _Float16* xr = lo16 + (size_t)(m0 + (lane & 15)) * KP_TAG;
    const _Float16* wr = wtag16 + (size_t)(lane & 15) * KP_TAG + hi * 16;

    v8f acc = {};
#pragma unroll
    for (int k = 0; k < 19; k++) {
        int k0 = k * 32;
        v16h a = mk16(*(const v8h*)(xr + k0 + hi * 16),
                      *(const v8h*)(xr + k0 + hi * 16 + 8));
        v16h b = mk16(*(const v8h*)(wr + k0),
                      *(const v8h*)(wr + k0 + 8));
        acc = __builtin_amdgcn_wmma_f32_16x16x32_f16(false, a, false, b,
                                                     (short)0, acc, false, false);
    }
    int n = lane & 15;
    float bb = (n < NTAGS) ? b_tag[n] : 0.f;
#pragma unroll
    for (int r = 0; r < 8; r++) {
        int m = m0 + r + 8 * hi;
        feats[(size_t)m * 16 + n] = acc[r] + bb;
    }
}

// ---------------------------------------------------------------------------
// Viterbi + backtrace: single block. feats and backpointers staged in LDS so
// the 2048-step dependent chain runs at LDS latency, not HBM latency.
// ---------------------------------------------------------------------------
__global__ __launch_bounds__(256) void viterbi(const float* __restrict__ feats,
                                               const float* __restrict__ trans,
                                               float* __restrict__ out) {
    __shared__ float fsh[T_LEN * NTAGS];          // 98304 B
    __shared__ unsigned char bph[T_LEN * NTAGS];  // 24576 B
    __shared__ float tr[NTAGS * NTAGS];
    __shared__ float fv[NTAGS];
    int tid = threadIdx.x;

    for (int i = tid; i < T_LEN * NTAGS; i += 256) {
        int t = i / NTAGS, n = i - t * NTAGS;
        fsh[i] = feats[t * 16 + n];
    }
    if (tid < NTAGS * NTAGS) tr[tid] = trans[tid];
    if (tid < NTAGS) fv[tid] = (tid == START_T) ? 0.f : NEGINF;
    __syncthreads();

    if (tid < 32) {                               // single-wave forward scan
        for (int t = 0; t < T_LEN; t++) {
            float best = NEGINF;
            int bp = 0;
            if (tid < NTAGS) {
                best = fv[0] + tr[tid * NTAGS];
#pragma unroll
                for (int p = 1; p < NTAGS; p++) {
                    float sc = fv[p] + tr[tid * NTAGS + p];
                    if (sc > best) { best = sc; bp = p; }
                }
                best += fsh[t * NTAGS + tid];
            }
            if (tid < NTAGS) {                    // in-wave DS ordering: loads precede stores
                fv[tid] = best;
                bph[t * NTAGS + tid] = (unsigned char)bp;
            }
        }
    }
    __syncthreads();

    if (tid == 0) {
        float best = -1e30f; int bt = 0;
        for (int n = 0; n < NTAGS; n++) {
            float sc = fv[n] + tr[STOP_T * NTAGS + n];
            if (sc > best) { best = sc; bt = n; }
        }
        out[0] = best;
        int tag = bt;
        for (int t = T_LEN - 1; t >= 0; t--) {    // emit then follow backpointer
            out[1 + t] = (float)tag;
            tag = bph[t * NTAGS + tag];
        }
    }
}

// ---------------------------------------------------------------------------
extern "C" void kernel_launch(void* const* d_in, const int* in_sizes, int n_in,
                              void* d_out, int out_size, void* d_ws, size_t ws_size,
                              hipStream_t stream) {
    const int*   indices = (const int*)d_in[0];
    const float* other   = (const float*)d_in[1];
    const float* emb     = (const float*)d_in[2];
    const float* Wi_f    = (const float*)d_in[3];
    const float* Wh_f    = (const float*)d_in[4];
    const float* bi_f    = (const float*)d_in[5];
    const float* bh_f    = (const float*)d_in[6];
    const float* Wi_b    = (const float*)d_in[7];
    const float* Wh_b    = (const float*)d_in[8];
    const float* bi_b    = (const float*)d_in[9];
    const float* bh_b    = (const float*)d_in[10];
    const float* W_tag   = (const float*)d_in[11];
    const float* b_tag   = (const float*)d_in[12];
    const float* trans   = (const float*)d_in[13];
    const float* h0      = (const float*)d_in[14];
    const float* c0      = (const float*)d_in[15];

    char* ws = (char*)d_ws;
    size_t off = 0;
    auto carve = [&](size_t bytes) -> char* {
        char* p = ws + off;
        off += (bytes + 255) & ~(size_t)255;
        return p;
    };
    _Float16* X16    = (_Float16*)carve((size_t)T_LEN * DIN * 2);       // 1.31 MB
    _Float16* Wi16   = (_Float16*)carve((size_t)2 * G4H * DIN * 2);     // 1.54 MB
    _Float16* Wh16   = (_Float16*)carve((size_t)2 * G4H * DIN * 2);     // 1.54 MB
    float*    bias   = (float*)   carve((size_t)2 * G4H * 4);
    float*    preact = (float*)   carve((size_t)2 * T_LEN * G4H * 4);   // 19.7 MB
    _Float16* lo16   = (_Float16*)carve((size_t)T_LEN * KP_TAG * 2);    // 2.49 MB
    _Float16* wtag16 = (_Float16*)carve((size_t)16 * KP_TAG * 2);
    float*    featsb = (float*)   carve((size_t)T_LEN * 16 * 4);

    pack_x   <<<T_LEN, DIN, 0, stream>>>(indices, other, emb, X16);
    pack_wi  <<<2 * G4H, DIN, 0, stream>>>(Wi_f, Wi_b, Wi16);
    pack_wh  <<<2 * G4H, DIN, 0, stream>>>(Wh_f, Wh_b, Wh16);
    pack_bias<<<10, 256, 0, stream>>>(bi_f, bh_f, bi_b, bh_b, bias);
    pack_wtag<<<16, KP_TAG, 0, stream>>>(W_tag, wtag16);
    zero_pad_lo<<<T_LEN, 8, 0, stream>>>(lo16);

    gemm_preact<<<dim3(T_LEN / 16, 19, 2), 128, 0, stream>>>(X16, Wi16, bias, preact);
    lstm_rec   <<<2, 512, 0, stream>>>(Wh16, preact, h0, c0, lo16);
    gemm_tag   <<<T_LEN / 64, 128, 0, stream>>>(lo16, wtag16, b_tag, featsb);
    viterbi    <<<1, 256, 0, stream>>>(featsb, trans, (float*)d_out);
}
